// MaskedMultiheadAttentionLayer_29411936043403
// MI455X (gfx1250) — compile-verified, matrix operand order checked by reading, not run on hardware
//
#include <hip/hip_runtime.h>
#include <hip/hip_bf16.h>
#include <math.h>

// ---------------------------------------------------------------------------
// MI455X (gfx1250) wave32 WMMA masked multi-head attention.
// B=2, S=2048, D=1024, H=16, DK=64.
// ---------------------------------------------------------------------------

#define BB 2
#define SS 2048
#define DD 1024
#define HH 16
#define DKK 64
#define QW 2            // query tiles (of 16) per attention wave; 2 waves/block

typedef __bf16 bf16;
typedef __attribute__((ext_vector_type(16))) bf16  bf16x16;
typedef __attribute__((ext_vector_type(8)))  bf16  bf16x8;
typedef __attribute__((ext_vector_type(4)))  bf16  bf16x4;
typedef __attribute__((ext_vector_type(8)))  float f32x8;

#define LDS_STRIDE 48   // GEMM tiles: 32 -> 48 elems (96B rows): aligned b128 reads
#define KSTRIDE    72   // attn K tile: 64 -> 72 elems (144B rows): conflict-free b128
#define VSTRIDE    48   // attn V tile: 32 -> 48 elems

__device__ __forceinline__ f32x8 wmma_bf16(bf16x16 a, bf16x16 b, f32x8 c) {
  return __builtin_amdgcn_wmma_f32_16x16x32_bf16(
      false, a, false, b, (short)0, c, false, false);
}

// 16x32 (MxK) bf16 fragment from a row-major global matrix.
// lanes 0-15  (row=lane)    : elems 0..7 = K+0..7,  8..15 = K+16..23
// lanes 16-31 (row=lane-16) : elems 0..7 = K+8..15, 8..15 = K+24..31
template <typename T>
__device__ __forceinline__ bf16x16 load_frag(const T* __restrict__ base, int ld,
                                             int row0, int k0) {
  const int lane = threadIdx.x & 31;
  const int r    = lane & 15;
  const int kh   = (lane >> 4) * 8;
  const T* p0 = base + (size_t)(row0 + r) * ld + (k0 + kh);
  const T* p1 = p0 + 16;
  bf16x16 f;
#pragma unroll
  for (int i = 0; i < 8; ++i) {
    f[i]     = (bf16)p0[i];
    f[i + 8] = (bf16)p1[i];
  }
  return f;
}

// Same fragment from an LDS tile with padded row stride (base includes row0/k0).
__device__ __forceinline__ bf16x16 lds_frag(const bf16* base, int stride) {
  const int lane = threadIdx.x & 31;
  const int row  = lane & 15;
  const int kh   = (lane >> 4) * 8;
  const bf16x8 lo = *(const bf16x8*)(base + row * stride + kh);
  const bf16x8 hi = *(const bf16x8*)(base + row * stride + kh + 16);
  bf16x16 f;
#pragma unroll
  for (int i = 0; i < 8; ++i) { f[i] = lo[i]; f[i + 8] = hi[i]; }
  return f;
}

template <typename T>
__device__ __forceinline__ void stage_read16(const T* __restrict__ p, float* buf) {
#pragma unroll
  for (int i = 0; i < 16; ++i) buf[i] = (float)p[i];
}

__device__ __forceinline__ void stage_write(bf16* dst, const float* buf, int nquad) {
#pragma unroll
  for (int i = 0; i < nquad; ++i) {
    bf16x4 q;
    q[0] = (bf16)buf[4 * i];     q[1] = (bf16)buf[4 * i + 1];
    q[2] = (bf16)buf[4 * i + 2]; q[3] = (bf16)buf[4 * i + 3];
    *(bf16x4*)(dst + 4 * i) = q;
  }
}

// ---------------------------------------------------------------------------
// GEMM: C[m][n] = sum_k A[m][k]*W[n][k] + bias[n]   (torch Linear)
// Block 256 thr (8 waves), tile 128M x 256N, K-step 32; wave tile 64x64.
// A/W staged through LDS (bf16, padded stride) with next-tile register
// prefetch overlapping the WMMA phase.
// MODE 0: bf16 [M][N]   MODE 1: bf16 Vt[b][h][dk][s]   MODE 2: fp32 [M][N]
// ---------------------------------------------------------------------------
template <typename AT, int MODE>
__global__ __launch_bounds__(256) void gemm_bias_kernel(
    const AT* __restrict__ A, const float* __restrict__ W,
    const float* __restrict__ bias, void* __restrict__ out,
    int M, int N, int K) {
  __shared__ bf16 ldsA[128 * LDS_STRIDE];   // 12.0 KB
  __shared__ bf16 ldsW[256 * LDS_STRIDE];   // 24.0 KB

  const int t    = threadIdx.x;
  const int wave = t >> 5;
  const int lane = t & 31;
  const int wm   = (wave & 1) * 64;
  const int wn   = (wave >> 1) * 64;
  const int mBlk = blockIdx.x * 128;
  const int nBlk = blockIdx.y * 256;

  const int arow = t >> 1;
  const int ach  = (t & 1) * 16;

  float abuf[16];
  float wbuf[32];

  stage_read16(A + (size_t)(mBlk + arow) * K + ach, abuf);
  stage_read16(W + (size_t)(nBlk + t) * K,          wbuf);
  stage_read16(W + (size_t)(nBlk + t) * K + 16,     wbuf + 16);

  f32x8 acc[4][4] = {};

  for (int k = 0; k < K; k += 32) {
    __syncthreads();
    stage_write(&ldsA[arow * LDS_STRIDE + ach], abuf, 4);
    stage_write(&ldsW[t * LDS_STRIDE],          wbuf, 8);
    if (k + 32 < K) {
      stage_read16(A + (size_t)(mBlk + arow) * K + (k + 32) + ach, abuf);
      stage_read16(W + (size_t)(nBlk + t) * K + (k + 32),          wbuf);
      stage_read16(W + (size_t)(nBlk + t) * K + (k + 32) + 16,     wbuf + 16);
    }
    __syncthreads();

    bf16x16 af[4], wf[4];
#pragma unroll
    for (int i = 0; i < 4; ++i)
      af[i] = lds_frag(ldsA + (wm + 16 * i) * LDS_STRIDE, LDS_STRIDE);
#pragma unroll
    for (int j = 0; j < 4; ++j)
      wf[j] = lds_frag(ldsW + (wn + 16 * j) * LDS_STRIDE, LDS_STRIDE);
#pragma unroll
    for (int i = 0; i < 4; ++i)
#pragma unroll
      for (int j = 0; j < 4; ++j)
        acc[i][j] = wmma_bf16(af[i], wf[j], acc[i][j]);
  }

  const int nlane = lane & 15;
  const int mh    = (lane >> 4) * 8;
#pragma unroll
  for (int i = 0; i < 4; ++i) {
#pragma unroll
    for (int j = 0; j < 4; ++j) {
      const int n = nBlk + wn + 16 * j + nlane;
      const float bv = bias[n];
      f32x8 c = acc[i][j];
#pragma unroll
      for (int r = 0; r < 8; ++r) {
        const int m = mBlk + wm + 16 * i + mh + r;
        const float v = c[r] + bv;
        if (MODE == 0) {
          ((bf16*)out)[(size_t)m * N + n] = (bf16)v;
        } else if (MODE == 1) {
          const int b_ = m >> 11, s = m & (SS - 1);
          const int h  = n >> 6,  dk = n & (DKK - 1);
          ((bf16*)out)[(((size_t)(b_ * HH + h)) * DKK + dk) * SS + s] = (bf16)v;
        } else {
          ((float*)out)[(size_t)m * N + n] = v;
        }
      }
    }
  }
}

// ---------------------------------------------------------------------------
// Flash attention, causal. Block = 64 threads (2 waves) per (batch, head,
// 64-query block); each wave owns 2 query tiles of 16. K (32x64) and Vt
// (64x32) blocks are staged in LDS once per key block and consumed by both
// waves, with register prefetch of the next block overlapping compute.
// Scores computed transposed (St = K_blk @ Q^T) so query = lane%16; exp'd P
// packs directly into the B-fragment of O^T = Vt @ P^T. Two-phase body
// (scores+softmax for both q-tiles, then V fragments loaded one at a time
// feeding both q-tiles) keeps peak VGPRs below 256.
// ---------------------------------------------------------------------------
__global__ __launch_bounds__(64) void attn_kernel(
    const bf16* __restrict__ Qb, const bf16* __restrict__ Kb,
    const bf16* __restrict__ Vt, bf16* __restrict__ attnOut) {
  __shared__ bf16 ldsK[32 * KSTRIDE];   // 4.5 KB
  __shared__ bf16 ldsV[64 * VSTRIDE];   // 6.0 KB

  const int q0blk = blockIdx.x * 64;
  const int h  = blockIdx.y;
  const int b  = blockIdx.z;
  const int t    = threadIdx.x;
  const int wave = t >> 5;
  const int lane = t & 31;
  const int qlane = lane & 15;
  const int kh8   = (lane >> 4) * 8;
  const int q0 = q0blk + wave * 32;          // this wave's 32 queries

  const bf16* Qh = Qb + (size_t)b * SS * DD + h * DKK;       // row stride D
  const bf16* Kh = Kb + (size_t)b * SS * DD + h * DKK;
  const bf16* Vh = Vt + ((size_t)(b * HH + h)) * DKK * SS;   // [DK][S]

  // Cooperative staging (64 threads): K tile 32x64 (2 thr/row, 64B each);
  // V tile 64x32 (1 thr/row, 64B each).
  const int krow = t >> 1;
  const int kcol = (t & 1) * 32;
  const bf16* kSrc = Kh + (size_t)krow * DD + kcol;
  const bf16* vSrc = Vh + (size_t)t * SS;

  bf16x16 qf[QW][2];
#pragma unroll
  for (int qt = 0; qt < QW; ++qt) {
    qf[qt][0] = load_frag(Qh, DD, q0 + 16 * qt, 0);
    qf[qt][1] = load_frag(Qh, DD, q0 + 16 * qt, 32);
  }

  f32x8 o[QW][4] = {};
  float mrun[QW], lrun[QW];
#pragma unroll
  for (int qt = 0; qt < QW; ++qt) { mrun[qt] = -INFINITY; lrun[qt] = 0.0f; }

  const float scale = 0.125f * 1.44269504088896340736f;  // 1/sqrt(DK)*log2(e)
  const int kmax = q0blk + 63;   // block-uniform causal bound

  // Prefetch key block kb=0 into registers.
  bf16x8 kreg[4], vreg[4];
#pragma unroll
  for (int i = 0; i < 4; ++i) {
    kreg[i] = *(const bf16x8*)(kSrc + 8 * i);
    vreg[i] = *(const bf16x8*)(vSrc + 8 * i);
  }

  for (int kb = 0; kb <= kmax; kb += 32) {
    __syncthreads();   // previous readers done with LDS
#pragma unroll
    for (int i = 0; i < 4; ++i) {
      *(bf16x8*)(&ldsK[krow * KSTRIDE + kcol + 8 * i]) = kreg[i];
      *(bf16x8*)(&ldsV[t * VSTRIDE + 8 * i])           = vreg[i];
    }
    if (kb + 32 <= kmax) {   // prefetch next block during compute
      const bf16* kn = kSrc + (size_t)(kb + 32) * DD;
      const bf16* vn = vSrc + (kb + 32);
#pragma unroll
      for (int i = 0; i < 4; ++i) {
        kreg[i] = *(const bf16x8*)(kn + 8 * i);
        vreg[i] = *(const bf16x8*)(vn + 8 * i);
      }
    }
    __syncthreads();   // tile visible

    // Phase 1: scores + softmax for both q-tiles (K fragments live here only).
    bf16x16 kf[2][2];
#pragma unroll
    for (int ktile = 0; ktile < 2; ++ktile)
#pragma unroll
      for (int c = 0; c < 2; ++c)
        kf[ktile][c] = lds_frag(ldsK + (ktile * 16) * KSTRIDE + c * 32, KSTRIDE);

    bf16x16 pf[QW];
    bool act[QW];
#pragma unroll
    for (int qt = 0; qt < QW; ++qt) {
      act[qt] = (kb <= q0 + 16 * qt + 15);   // wave-uniform causal skip
      if (act[qt]) {
        const int q_my = q0 + 16 * qt + qlane;
        f32x8 s0 = {}, s1 = {};
        s0 = wmma_bf16(kf[0][0], qf[qt][0], s0);
        s0 = wmma_bf16(kf[0][1], qf[qt][1], s0);
        s1 = wmma_bf16(kf[1][0], qf[qt][0], s1);
        s1 = wmma_bf16(kf[1][1], qf[qt][1], s1);

        float mx = -INFINITY;
#pragma unroll
        for (int r = 0; r < 8; ++r) {
          const int k0i = kb + kh8 + r;
          const int k1i = k0i + 16;
          s0[r] = (k0i <= q_my) ? s0[r] * scale : -INFINITY;
          s1[r] = (k1i <= q_my) ? s1[r] * scale : -INFINITY;
          mx = fmaxf(mx, fmaxf(s0[r], s1[r]));
        }
        mx = fmaxf(mx, __shfl_xor(mx, 16, 32));
        const float mnew  = fmaxf(mrun[qt], mx);
        const float alpha = exp2f(mrun[qt] - mnew);

        float rsum = 0.0f;
#pragma unroll
        for (int r = 0; r < 8; ++r) {
          const float p0 = exp2f(s0[r] - mnew);
          const float p1 = exp2f(s1[r] - mnew);
          rsum += p0 + p1;
          pf[qt][r]     = (bf16)p0;
          pf[qt][r + 8] = (bf16)p1;
        }
        rsum += __shfl_xor(rsum, 16, 32);
        lrun[qt] = lrun[qt] * alpha + rsum;
        mrun[qt] = mnew;

        // Rescale accumulators now so the PV phase is pure WMMA.
#pragma unroll
        for (int j = 0; j < 4; ++j)
#pragma unroll
          for (int r = 0; r < 8; ++r) o[qt][j][r] *= alpha;
      }
    }

    // Phase 2: O^T accumulate; each V fragment read from LDS once, used by
    // both q-tiles, then dead (only 8 VGPRs of V live at a time).
#pragma unroll
    for (int j = 0; j < 4; ++j) {
      const bf16x16 vf = lds_frag(ldsV + (j * 16) * VSTRIDE, VSTRIDE);
#pragma unroll
      for (int qt = 0; qt < QW; ++qt)
        if (act[qt]) o[qt][j] = wmma_bf16(vf, pf[qt], o[qt][j]);
    }
  }

  // Epilogue: O[q][d] = Ot[d][q]/l[q]; 16B packed bf16 stores.
#pragma unroll
  for (int qt = 0; qt < QW; ++qt) {
    const float rl = 1.0f / lrun[qt];
    bf16* orow = attnOut + ((size_t)(b * SS + q0 + 16 * qt + qlane)) * DD + h * DKK;
#pragma unroll
    for (int j = 0; j < 4; ++j) {
      bf16x8 pk;
#pragma unroll
      for (int r = 0; r < 8; ++r) pk[r] = (bf16)(o[qt][j][r] * rl);
      *(bf16x8*)(orow + j * 16 + kh8) = pk;
    }
  }
}

// ---------------------------------------------------------------------------
// Launch: 3 projections -> attention -> output projection.
// Workspace: Qb(8MB) | Kb(8MB) | Vt(8MB) | attn(8MB) = 32MB of d_ws.
// ---------------------------------------------------------------------------
extern "C" void kernel_launch(void* const* d_in, const int* in_sizes, int n_in,
                              void* d_out, int out_size, void* d_ws, size_t ws_size,
                              hipStream_t stream) {
  (void)in_sizes; (void)n_in; (void)out_size; (void)ws_size;
  const float* query = (const float*)d_in[0];
  const float* key   = (const float*)d_in[1];
  const float* value = (const float*)d_in[2];
  const float* wq = (const float*)d_in[3];
  const float* bq = (const float*)d_in[4];
  const float* wk = (const float*)d_in[5];
  const float* bk = (const float*)d_in[6];
  const float* wv = (const float*)d_in[7];
  const float* bv = (const float*)d_in[8];
  const float* wo = (const float*)d_in[9];
  const float* bo = (const float*)d_in[10];

  const size_t qkvBytes = (size_t)BB * SS * DD * sizeof(bf16);  // 8 MB
  char* ws   = (char*)d_ws;
  bf16* Qb   = (bf16*)(ws);
  bf16* Kb   = (bf16*)(ws + qkvBytes);
  bf16* VtB  = (bf16*)(ws + 2 * qkvBytes);
  bf16* attn = (bf16*)(ws + 3 * qkvBytes);

  const int M = BB * SS, N = DD, K = DD;
  dim3 gGrid(M / 128, N / 256);   // 32 x 4
  dim3 gBlk(256);

  gemm_bias_kernel<float, 0><<<gGrid, gBlk, 0, stream>>>(query, wq, bq, Qb,  M, N, K);
  gemm_bias_kernel<float, 0><<<gGrid, gBlk, 0, stream>>>(key,   wk, bk, Kb,  M, N, K);
  gemm_bias_kernel<float, 1><<<gGrid, gBlk, 0, stream>>>(value, wv, bv, VtB, M, N, K);

  dim3 aGrid(SS / 64, HH, BB);    // 32 x 16 x 2 blocks, 2 waves each
  attn_kernel<<<aGrid, dim3(64), 0, stream>>>(Qb, Kb, VtB, attn);

  gemm_bias_kernel<bf16, 2><<<gGrid, gBlk, 0, stream>>>(attn, wo, bo, (float*)d_out, M, N, K);
}
